// Attention_13314398617699
// MI455X (gfx1250) — matrix operand order repaired and checked by reading, not compiled, not run on hardware
//
#include <hip/hip_runtime.h>
#include <hip/hip_bf16.h>

// ---------------------------------------------------------------------------
// Attention layer for MI455X (gfx1250, wave32, WMMA).
// Pipeline:
//   1) f32->f16 convert: hidden, w_qkv, w_o
//   2) QKV GEMM (f16 WMMA, fp32 accum, 32x64 tile/wave, double-buffered K-steps)
//   3) RoPE in-place on Q,K heads of qkv_h (fp32 math)
//   4) V transpose -> vt_h [B][16][128][S] f16  (contiguous PV B-fragments)
//   5) Flash attention per (b,h,16-query tile): K frags pipelined into QK^T,
//      V frags issued after QK^T (live-range split, <256 VGPRs) and covered by
//      the softmax VALU section; online softmax; global_prefetch of next block
//   6) O-proj GEMM (f16 WMMA) -> fp32 d_out
// ---------------------------------------------------------------------------

typedef _Float16 v8h  __attribute__((ext_vector_type(8)));
typedef _Float16 v16h __attribute__((ext_vector_type(16)));
typedef float    v8f  __attribute__((ext_vector_type(8)));

#define BDIM 2
#define SDIM 2048
#define HID  2048
#define NH   16
#define NKV  16
#define HD   128
#define NHEADS (NH + 2*NKV)      // 48
#define QKVW (NHEADS * HD)       // 6144

__device__ __forceinline__ v16h pack16(v8h lo, v8h hi) {
  v16h r;
#pragma unroll
  for (int i = 0; i < 8; ++i) { r[i] = lo[i]; r[i + 8] = hi[i]; }
  return r;
}

// A-fragment (16x32 f16, MxK): lane L holds row L%16.
// K indices: i -> kb0+i, kb0 = (L<16?0:16).
__device__ __forceinline__ v16h load_a_frag(const _Float16* A, int lda, int row,
                                            int kk, int lane) {
  int kb0 = (lane < 16) ? 0 : 16;
  const _Float16* base = A + (size_t)row * lda + kk + kb0;
  v8h lo = *(const v8h*)(base);
  v8h hi = *(const v8h*)(base + 8);
  return pack16(lo, hi);
}

// ---------------------------------------------------------------------------
__global__ void cvt_f32_f16(const float* __restrict__ in,
                            _Float16* __restrict__ out, int n) {
  int i = blockIdx.x * blockDim.x + threadIdx.x;
  if (i < n) out[i] = (_Float16)in[i];
}

// ---------------------------------------------------------------------------
// C[M x N] = A[M x K] * W[N x K]^T, f16 inputs, fp32 accumulation.
// One wave per 32x64 tile; software-pipelined (double-buffered) K loop.
template <int OUT_F32>
__global__ __launch_bounds__(256)
void gemm32(const _Float16* __restrict__ A, const _Float16* __restrict__ W,
            void* __restrict__ Cout, int M, int N, int K,
            int lda, int ldw, int ldc) {
  int lane = threadIdx.x;
  int wid = blockIdx.x * blockDim.y + threadIdx.y;   // wave-uniform
  int nt4 = N >> 6;                                  // tiles of 64 cols
  int tm = wid / nt4;
  int tn = wid - tm * nt4;
  if (tm >= (M >> 5)) return;                        // uniform exit

  int row0 = tm * 32 + (lane & 15);
  int koff = (lane < 16) ? 0 : 16;
  v8f acc[2][4] = {};

  const _Float16* Wb = W + (size_t)(tn * 64 + (lane & 15)) * ldw + koff;

  // prologue: fragments for kk = 0
  v16h a0 = load_a_frag(A, lda, row0,      0, lane);
  v16h a1 = load_a_frag(A, lda, row0 + 16, 0, lane);
  v16h bf[4];
#pragma unroll
  for (int s = 0; s < 4; ++s)
    bf[s] = *(const v16h*)(Wb + (size_t)(s * 16) * ldw);

  for (int kk = 0; kk < K; kk += 32) {
    // fetch next K-step while current WMMAs execute
    v16h na0, na1, nbf[4];
    int kn = kk + 32;
    if (kn < K) {
      na0 = load_a_frag(A, lda, row0,      kn, lane);
      na1 = load_a_frag(A, lda, row0 + 16, kn, lane);
#pragma unroll
      for (int s = 0; s < 4; ++s)
        nbf[s] = *(const v16h*)(Wb + (size_t)(s * 16) * ldw + kn);
    }
#pragma unroll
    for (int s = 0; s < 4; ++s) {
      acc[0][s] = __builtin_amdgcn_wmma_f32_16x16x32_f16(
          false, a0, false, bf[s], (short)0, acc[0][s], false, false);
      acc[1][s] = __builtin_amdgcn_wmma_f32_16x16x32_f16(
          false, a1, false, bf[s], (short)0, acc[1][s], false, false);
    }
    a0 = na0; a1 = na1;
#pragma unroll
    for (int s = 0; s < 4; ++s) bf[s] = nbf[s];
  }

  int cbase = tn * 64 + (lane & 15);
#pragma unroll
  for (int r = 0; r < 2; ++r) {
    int rbase = tm * 32 + r * 16 + ((lane < 16) ? 0 : 8);
#pragma unroll
    for (int s = 0; s < 4; ++s) {
#pragma unroll
      for (int v = 0; v < 8; ++v) {
        size_t idx = (size_t)(rbase + v) * ldc + cbase + s * 16;
        if (OUT_F32) ((float*)Cout)[idx] = acc[r][s][v];
        else         ((_Float16*)Cout)[idx] = (_Float16)acc[r][s][v];
      }
    }
  }
}

// ---------------------------------------------------------------------------
// RoPE in place on Q (heads 0..15) and K (heads 16..31) of qkv_h.
__global__ void rope_kernel(_Float16* __restrict__ qkv,
                            const float* __restrict__ cosb,
                            const float* __restrict__ sinb) {
  size_t idx = (size_t)blockIdx.x * blockDim.x + threadIdx.x; // B*S*32*64
  int d    = idx & 63;
  int head = (idx >> 6) & 31;
  int s    = (idx >> 11) & (SDIM - 1);
  int b    = (int)(idx >> 22);
  _Float16* p = qkv + (((size_t)(b * SDIM + s)) * NHEADS + head) * HD;
  float x1 = (float)p[d];
  float x2 = (float)p[d + 64];
  float c1 = cosb[s * HD + d],      sn1 = sinb[s * HD + d];
  float c2 = cosb[s * HD + d + 64], sn2 = sinb[s * HD + d + 64];
  p[d]      = (_Float16)(x1 * c1 - x2 * sn1);   // q*cos + (-x2)*sin
  p[d + 64] = (_Float16)(x2 * c2 + x1 * sn2);   // q*cos + ( x1)*sin
}

// ---------------------------------------------------------------------------
// vt[b][h][d][s] = qkv[b][s][32+h][d]   (writes coalesced along s)
__global__ void vtrans_kernel(const _Float16* __restrict__ qkv,
                              _Float16* __restrict__ vt) {
  size_t idx = (size_t)blockIdx.x * blockDim.x + threadIdx.x; // B*16*128*S
  int s = idx & (SDIM - 1);
  int d = (idx >> 11) & (HD - 1);
  int h = (idx >> 18) & (NKV - 1);
  int b = (int)(idx >> 22);
  vt[idx] = qkv[(((size_t)(b * SDIM + s)) * NHEADS + 32 + h) * HD + d];
}

// ---------------------------------------------------------------------------
// Flash attention: one wave = one (b, h, 16-query tile). 4 waves / block.
__global__ __launch_bounds__(128)
void attn_fa(const _Float16* __restrict__ qkv, const _Float16* __restrict__ vt,
             _Float16* __restrict__ out) {
  __shared__ _Float16 plds[4][16][32];   // per-wave P staging (C-layout -> A-layout)
  int lane = threadIdx.x;
  int w    = threadIdx.y;
  int wid  = blockIdx.x * 4 + w;
  int qt = wid & (SDIM / 16 - 1);        // 128 query tiles
  int h  = (wid >> 7) & (NH - 1);
  int b  = wid >> 11;

  int kb0  = (lane < 16) ? 0 : 16;       // A-fragment K sub-base
  int koff = (lane < 16) ? 0 : 16;       // B-fragment K half

  // Q fragments for this tile (post-RoPE): 4 chunks of K=32 over HD=128
  int qrow = qt * 16 + (lane & 15);
  const _Float16* Qb = qkv + ((size_t)(b * SDIM + qrow)) * QKVW + h * HD;
  v16h qf[4];
#pragma unroll
  for (int c = 0; c < 4; ++c) {
    v8h lo = *(const v8h*)(Qb + c * 32 + kb0);
    v8h hi = *(const v8h*)(Qb + c * 32 + kb0 + 8);
    qf[c] = pack16(lo, hi);
  }

  const _Float16* Kh = qkv + ((size_t)b * SDIM) * QKVW + (NH + h) * HD;
  const _Float16* Vt = vt + ((size_t)(b * NKV + h)) * HD * SDIM;

  float m[8], l[8];
  v8f acc[8] = {};
#pragma unroll
  for (int v = 0; v < 8; ++v) { m[v] = -1e30f; l[v] = 0.f; }
  const float scale = 0.08838834764831845f;   // 1/sqrt(128)

  for (int kb = 0; kb < SDIM; kb += 32) {
    // ---- K fragments for this block (distinct registers -> pipelined loads) --
    v16h kf[8];                       // [c] keys kb..+15, [4+c] keys kb+16..+31
#pragma unroll
    for (int c = 0; c < 4; ++c) {
      kf[c]     = *(const v16h*)(Kh + (size_t)(kb +      (lane & 15)) * QKVW + c * 32 + koff);
      kf[4 + c] = *(const v16h*)(Kh + (size_t)(kb + 16 + (lane & 15)) * QKVW + c * 32 + koff);
    }

    // ---- S = Q K^T for 32 keys (two 16x16 score tiles) ----
    v8f s0 = {}, s1 = {};
#pragma unroll
    for (int c = 0; c < 4; ++c)
      s0 = __builtin_amdgcn_wmma_f32_16x16x32_f16(false, qf[c], false, kf[c],
                                                  (short)0, s0, false, false);
#pragma unroll
    for (int c = 0; c < 4; ++c)
      s1 = __builtin_amdgcn_wmma_f32_16x16x32_f16(false, qf[c], false, kf[4 + c],
                                                  (short)0, s1, false, false);

    // ---- V fragments: kf is dead now, reuse its registers; loads overlap the
    //      entire softmax VALU section below -------------------------------
    v16h vf[8];
#pragma unroll
    for (int d = 0; d < 8; ++d)
      vf[d] = *(const v16h*)(Vt + (size_t)(d * 16 + (lane & 15)) * SDIM + kb + koff);

    // prefetch next key block into cache
    if (kb + 32 < SDIM) {
      __builtin_prefetch(Kh + (size_t)(kb + 32 + (lane & 15)) * QKVW + koff, 0, 0);
      __builtin_prefetch(Kh + (size_t)(kb + 48 + (lane & 15)) * QKVW + koff, 0, 0);
      __builtin_prefetch(Vt + (size_t)(lane & 15) * SDIM + kb + 32 + koff, 0, 0);
    }

    // ---- online softmax update (row stats via 16-lane xor reductions) ----
#pragma unroll
    for (int v = 0; v < 8; ++v) {
      float a0 = s0[v] * scale, a1 = s1[v] * scale;
      float cm = fmaxf(a0, a1);
      cm = fmaxf(cm, __shfl_xor(cm, 1, 32));
      cm = fmaxf(cm, __shfl_xor(cm, 2, 32));
      cm = fmaxf(cm, __shfl_xor(cm, 4, 32));
      cm = fmaxf(cm, __shfl_xor(cm, 8, 32));
      float mn = fmaxf(m[v], cm);
      float corr = __expf(m[v] - mn);
      float p0 = __expf(a0 - mn);
      float p1 = __expf(a1 - mn);
      float rs = p0 + p1;
      rs += __shfl_xor(rs, 1, 32);
      rs += __shfl_xor(rs, 2, 32);
      rs += __shfl_xor(rs, 4, 32);
      rs += __shfl_xor(rs, 8, 32);
      l[v] = l[v] * corr + rs;
      m[v] = mn;
#pragma unroll
      for (int d = 0; d < 8; ++d) acc[d][v] *= corr;
      int r = v + ((lane < 16) ? 0 : 8);
      plds[w][r][lane & 15]        = (_Float16)p0;
      plds[w][r][16 + (lane & 15)] = (_Float16)p1;
    }
    asm volatile("s_wait_dscnt 0x0" ::: "memory");  // LDS stores visible to reads

    // ---- P (C-layout) -> A-fragment via LDS ----
    v8h plo = *(const v8h*)&plds[w][lane & 15][kb0];
    v8h phi = *(const v8h*)&plds[w][lane & 15][kb0 + 8];
    v16h pa = pack16(plo, phi);

    // ---- O += P V  (V fragments already resident) ----
#pragma unroll
    for (int d = 0; d < 8; ++d)
      acc[d] = __builtin_amdgcn_wmma_f32_16x16x32_f16(false, pa, false, vf[d],
                                                      (short)0, acc[d], false, false);
  }

  // ---- normalize and store attn [b][s][h*128+d] as f16 ----
#pragma unroll
  for (int v = 0; v < 8; ++v) {
    float inv = 1.0f / l[v];
    int r = qt * 16 + v + ((lane < 16) ? 0 : 8);
    _Float16* orow = out + ((size_t)(b * SDIM + r)) * (NH * HD) + h * HD;
#pragma unroll
    for (int d = 0; d < 8; ++d)
      orow[d * 16 + (lane & 15)] = (_Float16)(acc[d][v] * inv);
  }
}

// ---------------------------------------------------------------------------
extern "C" void kernel_launch(void* const* d_in, const int* in_sizes, int n_in,
                              void* d_out, int out_size, void* d_ws, size_t ws_size,
                              hipStream_t stream) {
  const float* hs   = (const float*)d_in[0];   // (B,S,HID)
  const float* cosb = (const float*)d_in[1];   // (S,1,HD)
  const float* sinb = (const float*)d_in[2];   // (S,1,HD)
  const float* wqkv = (const float*)d_in[3];   // (6144,2048)
  const float* wo   = (const float*)d_in[4];   // (2048,2048)
  float* outp = (float*)d_out;

  char* ws = (char*)d_ws;
  _Float16* hidden_h = (_Float16*)(ws);                          // 16 MB
  _Float16* wqkv_h   = (_Float16*)(ws + (size_t)16  * (1u<<20)); // 24 MB
  _Float16* wo_h     = (_Float16*)(ws + (size_t)40  * (1u<<20)); //  8 MB
  _Float16* qkv_h    = (_Float16*)(ws + (size_t)48  * (1u<<20)); // 48 MB
  _Float16* vt_h     = (_Float16*)(ws + (size_t)96  * (1u<<20)); // 16 MB
  _Float16* attn_h   = (_Float16*)(ws + (size_t)112 * (1u<<20)); // 16 MB

  const int M = BDIM * SDIM;  // 4096

  // 1) converts
  {
    int n0 = M * HID, n1 = QKVW * HID, n2 = HID * (NH * HD);
    cvt_f32_f16<<<(n0 + 255) / 256, 256, 0, stream>>>(hs, hidden_h, n0);
    cvt_f32_f16<<<(n1 + 255) / 256, 256, 0, stream>>>(wqkv, wqkv_h, n1);
    cvt_f32_f16<<<(n2 + 255) / 256, 256, 0, stream>>>(wo, wo_h, n2);
  }

  // 2) QKV GEMM: (4096x2048) x (6144x2048)^T -> f16 qkv_h
  {
    int waves = (M / 32) * (QKVW / 64);          // 128*96 = 12288
    gemm32<0><<<waves / 8, dim3(32, 8), 0, stream>>>(
        hidden_h, wqkv_h, qkv_h, M, QKVW, HID, HID, HID, QKVW);
  }

  // 3) RoPE in place on Q,K
  {
    size_t n = (size_t)BDIM * SDIM * 32 * 64;    // 8.4M pair-threads
    rope_kernel<<<(unsigned)(n / 256), 256, 0, stream>>>(qkv_h, cosb, sinb);
  }

  // 4) V transpose
  {
    size_t n = (size_t)BDIM * NKV * HD * SDIM;   // 8.4M
    vtrans_kernel<<<(unsigned)(n / 256), 256, 0, stream>>>(qkv_h, vt_h);
  }

  // 5) flash attention: 4096 waves, 4 waves/block
  attn_fa<<<1024, dim3(32, 4), 0, stream>>>(qkv_h, vt_h, attn_h);

  // 6) O-proj: (4096x2048) x (2048x2048)^T -> fp32 d_out
  {
    int waves = (M / 32) * (HID / 64);           // 128*32 = 4096
    gemm32<1><<<waves / 8, dim3(32, 8), 0, stream>>>(
        attn_h, wo_h, outp, M, HID, NH * HD, NH * HD, NH * HD, HID);
  }
}
